// Transformer_6365141532832
// MI455X (gfx1250) — matrix operand order repaired and checked by reading, not compiled, not run on hardware
//
#include <hip/hip_runtime.h>

// ---------------------------------------------------------------------------
// CDNA5 (gfx1250) WMMA types
// ---------------------------------------------------------------------------
typedef __attribute__((ext_vector_type(16))) __bf16 v16bf;
typedef __attribute__((ext_vector_type(8)))  float  v8f;

union FragH {
    v16bf  v;
    __bf16 h[16];
    uint4  q[2];
};

union Pack4 {
    __bf16 h[4];
    uint2  u;
};

__device__ __forceinline__ v8f wmma_bf16(v16bf a, v16bf b, v8f c) {
    return __builtin_amdgcn_wmma_f32_16x16x32_bf16(
        /*neg_a=*/false, a, /*neg_b=*/false, b,
        /*c_mod=*/(short)0, c, /*reuse_a=*/false, /*reuse_b=*/false);
}

// A-matrix fragment (16x32 bf16) from f32 memory row pointer p.
// Lane<16: K={0..7,16..23}; lane>=16: K={8..15,24..31}. klo = 0 or 8.
__device__ __forceinline__ FragH load_fragA_f32(const float* __restrict__ p, int klo) {
    FragH f;
    float4 a = *(const float4*)(p + klo);
    float4 b = *(const float4*)(p + klo + 4);
    float4 c = *(const float4*)(p + klo + 16);
    float4 d = *(const float4*)(p + klo + 20);
    f.h[0]=(__bf16)a.x;  f.h[1]=(__bf16)a.y;  f.h[2]=(__bf16)a.z;  f.h[3]=(__bf16)a.w;
    f.h[4]=(__bf16)b.x;  f.h[5]=(__bf16)b.y;  f.h[6]=(__bf16)b.z;  f.h[7]=(__bf16)b.w;
    f.h[8]=(__bf16)c.x;  f.h[9]=(__bf16)c.y;  f.h[10]=(__bf16)c.z; f.h[11]=(__bf16)c.w;
    f.h[12]=(__bf16)d.x; f.h[13]=(__bf16)d.y; f.h[14]=(__bf16)d.z; f.h[15]=(__bf16)d.w;
    return f;
}

// B-matrix fragment (32x16 bf16): lane = column;
// Lane<16: K={0..7,16..23}; lane>=16: K={8..15,24..31}. kb = 0 or 8.
__device__ __forceinline__ FragH load_fragB_f32(const float* __restrict__ p, int kb) {
    FragH f;
    float4 a = *(const float4*)(p + kb);
    float4 b = *(const float4*)(p + kb + 4);
    float4 c = *(const float4*)(p + kb + 16);
    float4 d = *(const float4*)(p + kb + 20);
    f.h[0]=(__bf16)a.x;  f.h[1]=(__bf16)a.y;  f.h[2]=(__bf16)a.z;  f.h[3]=(__bf16)a.w;
    f.h[4]=(__bf16)b.x;  f.h[5]=(__bf16)b.y;  f.h[6]=(__bf16)b.z;  f.h[7]=(__bf16)b.w;
    f.h[8]=(__bf16)c.x;  f.h[9]=(__bf16)c.y;  f.h[10]=(__bf16)c.z; f.h[11]=(__bf16)c.w;
    f.h[12]=(__bf16)d.x; f.h[13]=(__bf16)d.y; f.h[14]=(__bf16)d.z; f.h[15]=(__bf16)d.w;
    return f;
}

__device__ __forceinline__ float epilogue_op(float v, int act) {
    if (act == 1)      v = 0.5f * v * (1.0f + erff(v * 0.70710678118654752f));
    else if (act == 2) v = 1.0f / (1.0f + expf(-v));
    return v;
}

#define BM 128
#define BN 64
#define BK 32

// ---------------------------------------------------------------------------
// GEMM v2: C[M,N] = act(A[M,K] @ Bt^T + bias) + res, Bt is bf16 [N,K]
// (weights pre-converted+transposed). B tile staged to LDS with
// GLOBAL_LOAD_ASYNC_TO_LDS_B128 (ASYNCcnt), overlapped with f32->bf16 A staging.
// N % 64 == 0, K % 32 == 0. act: 0 none, 1 exact GELU, 2 sigmoid.
// ---------------------------------------------------------------------------
__launch_bounds__(256)
__global__ void gemm_bt_async_kernel(const float* __restrict__ A,
                                     const __bf16* __restrict__ Bt,
                                     const float* __restrict__ bias,
                                     const float* __restrict__ res,
                                     float* __restrict__ C,
                                     int M, int N, int K, int act)
{
    __shared__ __bf16 As[BM * BK];   // 8 KB, [m][k]
    __shared__ __bf16 Bs[BN * BK];   // 4 KB, [n][k]

    const int t    = threadIdx.x;
    const int lane = t & 31;
    const int wave = t >> 5;
    const int wr   = wave >> 1;
    const int wc   = wave & 1;
    const int mBase = blockIdx.y * BM;
    const int nBase = blockIdx.x * BN;
    const int lm   = lane & 15;
    const int kloA = (lane < 16) ? 0 : 8;
    const int kbB  = (lane < 16) ? 0 : 16;

    // per-thread async-copy slot: 16 bytes of the [n][k] B tile
    const int bn = t >> 2;            // 0..63 (output column within tile)
    const int bc = (t & 3) * 8;       // 8 bf16 = 16 bytes along K
    const unsigned ldsB = (unsigned)(uintptr_t)&Bs[bn * BK + bc];
    const __bf16* gB = Bt + (size_t)(nBase + bn) * K + bc;

    v8f acc[2][2] = {};

    for (int k0 = 0; k0 < K; k0 += BK) {
        // ---- B tile: async DMA, global bf16 -> LDS, no VGPR round trip ----
        {
            unsigned long long ga = (unsigned long long)(uintptr_t)(gB + k0);
            asm volatile("global_load_async_to_lds_b128 %0, %1, off"
                         :: "v"(ldsB), "v"(ga) : "memory");
        }
        // ---- A tile: 128x32 f32 -> bf16 (overlaps with the async DMA) ----
        #pragma unroll
        for (int it = 0; it < 4; ++it) {
            int idx = t + it * 256;
            int row = idx >> 3;
            int cv  = (idx & 7) << 2;
            int gr  = mBase + row;
            int grc = (gr < M) ? gr : (M - 1);
            float4 f = *(const float4*)(A + (size_t)grc * K + k0 + cv);
            Pack4 p4;
            p4.h[0]=(__bf16)f.x; p4.h[1]=(__bf16)f.y;
            p4.h[2]=(__bf16)f.z; p4.h[3]=(__bf16)f.w;
            *(uint2*)&As[row * BK + cv] = p4.u;
        }
        asm volatile("s_wait_asynccnt 0x0" ::: "memory");
        __syncthreads();

        FragH a[2], b[2];
        #pragma unroll
        for (int i = 0; i < 2; ++i) {
            int row = wr * 32 + i * 16 + lm;
            a[i].q[0] = *(const uint4*)&As[row * BK + kloA];
            a[i].q[1] = *(const uint4*)&As[row * BK + kloA + 16];
        }
        #pragma unroll
        for (int j = 0; j < 2; ++j) {
            int col = wc * 32 + j * 16 + lm;
            b[j].q[0] = *(const uint4*)&Bs[col * BK + kloA];
            b[j].q[1] = *(const uint4*)&Bs[col * BK + kloA + 16];
        }
        #pragma unroll
        for (int i = 0; i < 2; ++i)
            #pragma unroll
            for (int j = 0; j < 2; ++j)
                acc[i][j] = wmma_bf16(a[i].v, b[j].v, acc[i][j]);
        __syncthreads();
    }

    const int rsel = (lane >> 4) * 8;
    #pragma unroll
    for (int i = 0; i < 2; ++i) {
        #pragma unroll
        for (int j = 0; j < 2; ++j) {
            int col = nBase + wc * 32 + j * 16 + lm;
            float bv = bias ? bias[col] : 0.0f;
            #pragma unroll
            for (int r = 0; r < 8; ++r) {
                int row = mBase + wr * 32 + i * 16 + rsel + r;
                if (row < M) {
                    float v = epilogue_op(acc[i][j][r] + bv, act);
                    if (res) v += res[(size_t)row * N + col];
                    C[(size_t)row * N + col] = v;
                }
            }
        }
    }
}

// ---------------------------------------------------------------------------
// GEMM (f32 B path) — used only for the tiny gate-FF GEMMs (M=2), whose
// weights are read exactly once (conversion would cost more than it saves).
// ---------------------------------------------------------------------------
__launch_bounds__(256)
__global__ void gemm_f32_kernel(const float* __restrict__ A,
                                const float* __restrict__ B,
                                const float* __restrict__ bias,
                                const float* __restrict__ res,
                                float* __restrict__ C,
                                int M, int N, int K, int act)
{
    __shared__ __bf16 As[BM * BK];
    __shared__ __bf16 Bs[BN * BK];   // transposed [n][k]

    const int t    = threadIdx.x;
    const int lane = t & 31;
    const int wave = t >> 5;
    const int wr   = wave >> 1;
    const int wc   = wave & 1;
    const int mBase = blockIdx.y * BM;
    const int nBase = blockIdx.x * BN;
    const int lm   = lane & 15;
    const int kloA = (lane < 16) ? 0 : 8;
    const int kbB  = (lane < 16) ? 0 : 16;

    v8f acc[2][2] = {};

    for (int k0 = 0; k0 < K; k0 += BK) {
        #pragma unroll
        for (int it = 0; it < 4; ++it) {
            int idx = t + it * 256;
            int row = idx >> 3;
            int cv  = (idx & 7) << 2;
            int gr  = mBase + row;
            int grc = (gr < M) ? gr : (M - 1);
            float4 f = *(const float4*)(A + (size_t)grc * K + k0 + cv);
            Pack4 p4;
            p4.h[0]=(__bf16)f.x; p4.h[1]=(__bf16)f.y;
            p4.h[2]=(__bf16)f.z; p4.h[3]=(__bf16)f.w;
            *(uint2*)&As[row * BK + cv] = p4.u;
        }
        #pragma unroll
        for (int it = 0; it < 8; ++it) {
            int idx = t + it * 256;
            int n = idx & 63;
            int k = idx >> 6;
            Bs[n * BK + k] = (__bf16)B[(size_t)(k0 + k) * N + nBase + n];
        }
        __syncthreads();

        FragH a[2], b[2];
        #pragma unroll
        for (int i = 0; i < 2; ++i) {
            int row = wr * 32 + i * 16 + lm;
            a[i].q[0] = *(const uint4*)&As[row * BK + kloA];
            a[i].q[1] = *(const uint4*)&As[row * BK + kloA + 16];
        }
        #pragma unroll
        for (int j = 0; j < 2; ++j) {
            int col = wc * 32 + j * 16 + lm;
            b[j].q[0] = *(const uint4*)&Bs[col * BK + kloA];
            b[j].q[1] = *(const uint4*)&Bs[col * BK + kloA + 16];
        }
        #pragma unroll
        for (int i = 0; i < 2; ++i)
            #pragma unroll
            for (int j = 0; j < 2; ++j)
                acc[i][j] = wmma_bf16(a[i].v, b[j].v, acc[i][j]);
        __syncthreads();
    }

    const int rsel = (lane >> 4) * 8;
    #pragma unroll
    for (int i = 0; i < 2; ++i) {
        #pragma unroll
        for (int j = 0; j < 2; ++j) {
            int col = nBase + wc * 32 + j * 16 + lm;
            float bv = bias ? bias[col] : 0.0f;
            #pragma unroll
            for (int r = 0; r < 8; ++r) {
                int row = mBase + wr * 32 + i * 16 + rsel + r;
                if (row < M) {
                    float v = epilogue_op(acc[i][j][r] + bv, act);
                    if (res) v += res[(size_t)row * N + col];
                    C[(size_t)row * N + col] = v;
                }
            }
        }
    }
}

// ---------------------------------------------------------------------------
// Weight convert + transpose: W[K,N] f32 -> Wt[N,K] bf16 (LDS-tiled, padded).
// K, N multiples of 32.
// ---------------------------------------------------------------------------
__launch_bounds__(256)
__global__ void cvt_transpose_kernel(const float* __restrict__ W,
                                     __bf16* __restrict__ Wt, int K, int N)
{
    __shared__ float tile[32][33];
    const int n0 = blockIdx.x * 32;
    const int k0 = blockIdx.y * 32;
    const int tx = threadIdx.x & 31;
    const int ty = threadIdx.x >> 5;
    for (int r = ty; r < 32; r += 8)
        tile[r][tx] = W[(size_t)(k0 + r) * N + n0 + tx];
    __syncthreads();
    for (int r = ty; r < 32; r += 8)
        Wt[(size_t)(n0 + r) * K + k0 + tx] = (__bf16)tile[tx][r];
}

// ---------------------------------------------------------------------------
// Flash attention, head dim 64, one wave per (query-block-of-16, head, batch).
// ---------------------------------------------------------------------------
__launch_bounds__(32)
__global__ void attn_flash_kernel(const float* __restrict__ q,
                                  const float* __restrict__ k,
                                  const float* __restrict__ v,
                                  float* __restrict__ out,
                                  int nq, int nk,
                                  int qRow, int kRow, int vRow, int oRow,
                                  int qCol, int kCol, int vCol, int oCol,
                                  long qB, long kB, long vB, long oB)
{
    __shared__ float  Stile[16 * 33];
    __shared__ __bf16 Ptile[16 * 32];
    __shared__ __bf16 Vt[64 * 32];
    __shared__ float  alphaS[16];
    __shared__ float  lS[16];

    const int lane  = threadIdx.x;
    const int head  = blockIdx.y;
    const int bb    = blockIdx.z;
    const int qbase = blockIdx.x * 16;
    const int lm    = lane & 15;
    const int kloA  = (lane < 16) ? 0 : 8;
    const int kbB   = (lane < 16) ? 0 : 16;
    const int rsel  = (lane >> 4) * 8;

    const float* qp = q + bb * qB + qCol + head * 64;
    const float* kp = k + bb * kB + kCol + head * 64;
    const float* vp = v + bb * vB + vCol + head * 64;
    float*       op = out + bb * oB + oCol + head * 64;

    FragH aQ[2];
    {
        int row = qbase + lm;
        if (row >= nq) row = nq - 1;
        const float* qr = qp + (size_t)row * qRow;
        aQ[0] = load_fragA_f32(qr, kloA);
        aQ[1] = load_fragA_f32(qr + 32, kloA);
    }

    v8f oAcc[4] = {};
    float m = -3.0e38f, l = 0.0f;
    const float scale = 0.125f;   // 64^-0.5

    for (int kv0 = 0; kv0 < nk; kv0 += 32) {
        v8f s[2] = {};
        #pragma unroll
        for (int ct = 0; ct < 2; ++ct) {
            int krow = kv0 + ct * 16 + lm;
            if (krow >= nk) krow = nk - 1;
            const float* kr = kp + (size_t)krow * kRow;
            FragH bK0 = load_fragB_f32(kr, kloA);
            FragH bK1 = load_fragB_f32(kr + 32, kloA);
            s[ct] = wmma_bf16(aQ[0].v, bK0.v, s[ct]);
            s[ct] = wmma_bf16(aQ[1].v, bK1.v, s[ct]);
        }
        #pragma unroll
        for (int ct = 0; ct < 2; ++ct)
            #pragma unroll
            for (int r = 0; r < 8; ++r)
                Stile[(rsel + r) * 33 + ct * 16 + lm] = s[ct][r] * scale;
        __syncthreads();

        int kcnt = nk - kv0; if (kcnt > 32) kcnt = 32;
        if (lane < 16) {
            const float* Srow = &Stile[lane * 33];
            float rowmax = -3.0e38f;
            for (int j = 0; j < kcnt; ++j) rowmax = fmaxf(rowmax, Srow[j]);
            float mnew  = fmaxf(m, rowmax);
            float alpha = expf(m - mnew);
            float psum  = 0.0f;
            for (int j = 0; j < 32; ++j) {
                float pj = (j < kcnt) ? expf(Srow[j] - mnew) : 0.0f;
                Ptile[lane * 32 + j] = (__bf16)pj;
                psum += pj;
            }
            l = l * alpha + psum;
            m = mnew;
            alphaS[lane] = alpha;
        }

        #pragma unroll
        for (int it = 0; it < 16; ++it) {
            int idx = it * 32 + lane;
            int kvr = idx >> 4;
            int dv  = (idx & 15) * 4;
            int gkv = kv0 + kvr;
            if (gkv >= nk) gkv = nk - 1;
            float4 f = *(const float4*)(vp + (size_t)gkv * vRow + dv);
            Vt[(dv + 0) * 32 + kvr] = (__bf16)f.x;
            Vt[(dv + 1) * 32 + kvr] = (__bf16)f.y;
            Vt[(dv + 2) * 32 + kvr] = (__bf16)f.z;
            Vt[(dv + 3) * 32 + kvr] = (__bf16)f.w;
        }
        __syncthreads();

        float al[8];
        #pragma unroll
        for (int r = 0; r < 8; ++r) al[r] = alphaS[rsel + r];
        #pragma unroll
        for (int f4 = 0; f4 < 4; ++f4)
            #pragma unroll
            for (int r = 0; r < 8; ++r)
                oAcc[f4][r] *= al[r];

        FragH aP;
        aP.q[0] = *(const uint4*)&Ptile[lm * 32 + kloA];
        aP.q[1] = *(const uint4*)&Ptile[lm * 32 + kloA + 16];
        #pragma unroll
        for (int f4 = 0; f4 < 4; ++f4) {
            int col = f4 * 16 + lm;
            FragH bV;
            bV.q[0] = *(const uint4*)&Vt[col * 32 + kloA];
            bV.q[1] = *(const uint4*)&Vt[col * 32 + kloA + 16];
            oAcc[f4] = wmma_bf16(aP.v, bV.v, oAcc[f4]);
        }
        __syncthreads();
    }

    if (lane < 16) lS[lane] = l;
    __syncthreads();

    float invl[8];
    #pragma unroll
    for (int r = 0; r < 8; ++r) invl[r] = 1.0f / lS[rsel + r];
    #pragma unroll
    for (int f4 = 0; f4 < 4; ++f4) {
        int col = f4 * 16 + lm;
        #pragma unroll
        for (int r = 0; r < 8; ++r) {
            int row = qbase + rsel + r;
            if (row < nq)
                op[(size_t)row * oRow + col] = oAcc[f4][r] * invl[r];
        }
    }
}

// ---------------------------------------------------------------------------
// LayerNorm over last dim W, one block (256 threads) per row.
// ---------------------------------------------------------------------------
__launch_bounds__(256)
__global__ void layernorm_kernel(const float* __restrict__ x,
                                 const float* __restrict__ g,
                                 const float* __restrict__ b,
                                 float* __restrict__ y, int W)
{
    const int row = blockIdx.x;
    const float* xr = x + (size_t)row * W;
    float* yr = y + (size_t)row * W;
    __shared__ float red[256];
    const int t = threadIdx.x;

    float s = 0.0f;
    for (int i = t; i < W; i += 256) s += xr[i];
    red[t] = s; __syncthreads();
    for (int off = 128; off > 0; off >>= 1) {
        if (t < off) red[t] += red[t + off];
        __syncthreads();
    }
    float mean = red[0] / W;
    __syncthreads();

    float s2 = 0.0f;
    for (int i = t; i < W; i += 256) { float d = xr[i] - mean; s2 += d * d; }
    red[t] = s2; __syncthreads();
    for (int off = 128; off > 0; off >>= 1) {
        if (t < off) red[t] += red[t + off];
        __syncthreads();
    }
    float rstd = rsqrtf(red[0] / W + 1e-5f);

    for (int i = t; i < W; i += 256)
        yr[i] = (xr[i] - mean) * rstd * g[i] + b[i];
}

__launch_bounds__(256)
__global__ void meanpool_kernel(const float* __restrict__ x, float* __restrict__ pooled,
                                int n, int Cc, int off, int DIMt)
{
    int c = blockIdx.x * 256 + threadIdx.x;
    int bb = blockIdx.y;
    if (c >= Cc) return;
    const float* xb = x + (size_t)bb * n * Cc;
    float s = 0.0f;
    for (int t = 0; t < n; ++t) s += xb[(size_t)t * Cc + c];
    pooled[(size_t)bb * DIMt + off + c] = s / (float)n;
}

__launch_bounds__(256)
__global__ void gate_apply_kernel(float* __restrict__ x, const float* __restrict__ s,
                                  int n, int Cc, int off, int DIMt, long total)
{
    long idx = (long)blockIdx.x * 256 + threadIdx.x;
    if (idx >= total) return;
    int c  = (int)(idx % Cc);
    int bb = (int)(idx / ((long)n * Cc));
    x[idx] *= (1.0f + s[(size_t)bb * DIMt + off + c]);
}

// ---------------------------------------------------------------------------
// Host orchestration
// ---------------------------------------------------------------------------
static inline void launch_gemm_bt(hipStream_t st, const float* A, const __bf16* Bt,
                                  const float* bias, const float* res, float* C,
                                  int M, int N, int K, int act)
{
    dim3 g(N / 64, (M + BM - 1) / BM);
    gemm_bt_async_kernel<<<g, 256, 0, st>>>(A, Bt, bias, res, C, M, N, K, act);
}

static inline void launch_gemm_f32(hipStream_t st, const float* A, const float* B,
                                   const float* bias, const float* res, float* C,
                                   int M, int N, int K, int act)
{
    dim3 g(N / 64, (M + BM - 1) / BM);
    gemm_f32_kernel<<<g, 256, 0, st>>>(A, B, bias, res, C, M, N, K, act);
}

static inline void launch_cvtT(hipStream_t st, const float* W, __bf16* Wt, int K, int N)
{
    dim3 g(N / 32, K / 32);
    cvt_transpose_kernel<<<g, 256, 0, st>>>(W, Wt, K, N);
}

extern "C" void kernel_launch(void* const* d_in, const int* in_sizes, int n_in,
                              void* d_out, int out_size, void* d_ws, size_t ws_size,
                              hipStream_t stream)
{
    (void)in_sizes; (void)n_in; (void)out_size; (void)ws_size;

    const int Ns[3] = {196, 784, 3136};
    const int Bn = 2, Cd = 768;
    const int Ms[3] = {Bn * Ns[0], Bn * Ns[1], Bn * Ns[2]};

    auto P = [&](int i) { return (const float*)d_in[i]; };
    // d_in (JAX pytree order): 0:x1 1:x2 2:x3 3:final_b 4:final_g,
    // per layer (38): attn Wo0-2(+0) Wqkv0-2(+3) bo0-2(+6);
    // cross Wk0-2(+9) Wo0-2(+12) Wq0-2(+15) Wv0-2(+18) b(+21) bo0-2(+22) g(+25);
    // ff W1(+26) W2(+27) b(+28) b1(+29) b2(+30) g(+31);
    // sff W1(+32) W2(+33) b(+34) b1(+35) b2(+36) g(+37).

    float* w = (float*)d_ws;
    size_t off = 0;
    auto alloc = [&](size_t n) { float* p = w + off; off += n; return p; };

    float* cur[3]; for (int i = 0; i < 3; ++i) cur[i] = alloc((size_t)Ms[i] * Cd);
    float* L[3];   for (int i = 0; i < 3; ++i) L[i]   = alloc((size_t)Ms[i] * Cd);
    float* Obuf  = alloc((size_t)Ms[2] * 256);
    float* Hbuf  = alloc((size_t)1568 * 3072);
    float *qC[3], *kC[3], *vC[3];
    for (int i = 0; i < 3; ++i) {
        qC[i] = alloc((size_t)Ms[i] * 192);
        kC[i] = alloc((size_t)Ms[i] * 192);
        vC[i] = alloc((size_t)Ms[i] * 192);
    }
    float* Ccat   = alloc((size_t)Ms[2] * 384);
    float* pooled = alloc((size_t)Bn * 2304);
    float* pln    = alloc((size_t)Bn * 2304);
    float* gs     = alloc((size_t)Bn * 2304);
    // bf16 weight staging slots (2,359,296 bf16 each = W1t/W2t exactly)
    __bf16* wA = (__bf16*)alloc(1179648);
    __bf16* wB = (__bf16*)alloc(1179648);

    for (int i = 0; i < 3; ++i)
        hipMemcpyAsync(cur[i], d_in[i], sizeof(float) * (size_t)Ms[i] * Cd,
                       hipMemcpyDeviceToDevice, stream);

    for (int layer = 0; layer < 2; ++layer) {
        const int bi = 5 + layer * 38;

        // ---------------- self attention (x not normed, per reference) ------
        for (int i = 0; i < 3; ++i) {
            int M = Ms[i], n = Ns[i];
            launch_cvtT(stream, P(bi + 3 + i), wA, 768, 768);   // Wqkv^T
            launch_cvtT(stream, P(bi + 0 + i), wB, 256, 768);   // Wo^T
            launch_gemm_bt(stream, cur[i], wA, nullptr, nullptr, L[i], M, 768, 768, 0);
            dim3 ag((n + 15) / 16, 4, 2);
            attn_flash_kernel<<<ag, 32, 0, stream>>>(
                L[i], L[i], L[i], Obuf, n, n,
                768, 768, 768, 256, 0, 256, 512, 0,
                (long)n * 768, (long)n * 768, (long)n * 768, (long)n * 256);
            launch_gemm_bt(stream, Obuf, wB, P(bi + 6 + i), cur[i], cur[i],
                           M, 768, 256, 0);
        }

        // ---------------- split feed-forward (weights shared across scales) -
        launch_cvtT(stream, P(bi + 32), wA, 768, 3072);   // W1^T
        launch_cvtT(stream, P(bi + 33), wB, 3072, 768);   // W2^T
        for (int i = 0; i < 3; ++i) {
            int M = Ms[i];
            layernorm_kernel<<<M, 256, 0, stream>>>(cur[i], P(bi + 37), P(bi + 34), L[i], 768);
            for (int r0 = 0; r0 < M; r0 += 1568) {
                int mc = (M - r0 < 1568) ? (M - r0) : 1568;
                launch_gemm_bt(stream, L[i] + (size_t)r0 * 768, wA, P(bi + 35),
                               nullptr, Hbuf, mc, 3072, 768, 1);
                launch_gemm_bt(stream, Hbuf, wB, P(bi + 36),
                               cur[i] + (size_t)r0 * 768, cur[i] + (size_t)r0 * 768,
                               mc, 768, 3072, 1);
            }
        }

        // ---------------- cross attention ----------------------------------
        for (int i = 0; i < 3; ++i)
            layernorm_kernel<<<Ms[i], 256, 0, stream>>>(cur[i], P(bi + 25), P(bi + 21), L[i], 768);
        for (int i = 0; i < 3; ++i) {
            __bf16* wq = wA;
            __bf16* wk = wA + 768 * 192;
            __bf16* wv = wA + 2 * 768 * 192;
            launch_cvtT(stream, P(bi + 15 + i), wq, 768, 192);
            launch_cvtT(stream, P(bi + 9 + i),  wk, 768, 192);
            launch_cvtT(stream, P(bi + 18 + i), wv, 768, 192);
            launch_gemm_bt(stream, L[i], wq, nullptr, nullptr, qC[i], Ms[i], 192, 768, 0);
            launch_gemm_bt(stream, L[i], wk, nullptr, nullptr, kC[i], Ms[i], 192, 768, 0);
            launch_gemm_bt(stream, L[i], wv, nullptr, nullptr, vC[i], Ms[i], 192, 768, 0);
        }
        const int pair[3][2] = {{1, 2}, {0, 2}, {0, 1}};
        for (int i = 0; i < 3; ++i) {
            int nq = Ns[i];
            for (int p2 = 0; p2 < 2; ++p2) {
                int j = pair[i][p2];
                int nk = Ns[j];
                dim3 ag((nq + 15) / 16, 3, 2);
                attn_flash_kernel<<<ag, 32, 0, stream>>>(
                    qC[i], kC[j], vC[j], Ccat, nq, nk,
                    192, 192, 192, 384, 0, 0, 0, p2 * 192,
                    (long)nq * 192, (long)nk * 192, (long)nk * 192, (long)nq * 384);
            }
            launch_cvtT(stream, P(bi + 12 + i), wB, 384, 768);  // cross Wo^T
            // cross output has NO residual (replaces xs) per reference
            launch_gemm_bt(stream, Ccat, wB, P(bi + 22 + i), nullptr, cur[i],
                           Ms[i], 768, 384, 0);
        }

        // ---------------- gate feed-forward (M=2: keep f32-B path) ----------
        for (int i = 0; i < 3; ++i)
            meanpool_kernel<<<dim3(3, 2), 256, 0, stream>>>(cur[i], pooled, Ns[i], 768,
                                                            i * 768, 2304);
        layernorm_kernel<<<2, 256, 0, stream>>>(pooled, P(bi + 31), P(bi + 28), pln, 2304);
        launch_gemm_f32(stream, pln,  P(bi + 26), P(bi + 29), nullptr, Hbuf, 2, 9216, 2304, 1);
        launch_gemm_f32(stream, Hbuf, P(bi + 27), P(bi + 30), nullptr, gs,   2, 2304, 9216, 2);
        for (int i = 0; i < 3; ++i) {
            long total = (long)Ms[i] * 768;
            gate_apply_kernel<<<(unsigned)((total + 255) / 256), 256, 0, stream>>>(
                cur[i], gs, Ns[i], 768, i * 768, 2304, total);
        }
    }

    // ---------------- final layernorm -> d_out ------------------------------
    float* outp = (float*)d_out;
    size_t oo = 0;
    for (int i = 0; i < 3; ++i) {
        layernorm_kernel<<<Ms[i], 256, 0, stream>>>(cur[i], P(4), P(3), outp + oo, 768);
        oo += (size_t)Ms[i] * Cd;
    }
}
